// TensorComposerMobius_90520730730863
// MI455X (gfx1250) — matrix operand order repaired and checked by reading, not compiled, not run on hardware
//
#include <hip/hip_runtime.h>
#include <hip/hip_bf16.h>

typedef __attribute__((ext_vector_type(16))) _Float16 v16h;
typedef __attribute__((ext_vector_type(8)))  _Float16 v8h;
typedef __attribute__((ext_vector_type(8)))  float    v8f;

#define B_   8
#define CTX_ 2048
#define L_   256
#define K_   16
#define C_   16
#define R_   16
#define RT_  8
#define T_   18
#define O0_  256
#define O1_  256
#define EPS_ 1e-12f

__device__ __forceinline__ float siglog_f(float x) {
    float s = (x > 0.f) ? 1.f : ((x < 0.f) ? -1.f : 0.f);
    return s * log1pf(fabsf(x));
}

// ---------------------------------------------------------------------------
// Kernel 1: mod[b, j] = h1 * siglog(h2), h1 = x[b]·ctx_w[j]+b[j],
//           h2 = x[b]·ctx_w[j+2L]+b[j+2L].  mod flat == [B, L, 2].
// ---------------------------------------------------------------------------
__global__ void mod_kernel(const float* __restrict__ x, const float* __restrict__ w,
                           const float* __restrict__ bias, float* __restrict__ mod) {
    int j = blockIdx.x;        // 0..511 (= 2L)
    int b = blockIdx.y;        // 0..7
    int tid = threadIdx.x;
    const float* xr = x + (size_t)b * CTX_;
    const float* w1 = w + (size_t)j * CTX_;
    const float* w2 = w + (size_t)(j + 2 * L_) * CTX_;
    float a1 = 0.f, a2 = 0.f;
    for (int k = tid; k < CTX_; k += 256) {
        float xv = xr[k];
        a1 += xv * w1[k];
        a2 += xv * w2[k];
    }
    __shared__ float s1[256], s2[256];
    s1[tid] = a1; s2[tid] = a2;
    __syncthreads();
    for (int st = 128; st > 0; st >>= 1) {
        if (tid < st) { s1[tid] += s1[tid + st]; s2[tid] += s2[tid + st]; }
        __syncthreads();
    }
    if (tid == 0) {
        float h1 = s1[0] + bias[j];
        float h2 = s2[0] + bias[j + 2 * L_];
        mod[(size_t)b * 2 * L_ + j] = h1 * siglog_f(h2);
    }
}

// ---------------------------------------------------------------------------
// Kernel 2: Ck[t,c,l,r] = sum_k cmul(A[t,c,l,k,r], B[t,c,k,l,r]).
// t<2 -> store full Ck01[t,c,l,r,2]; t>=2 -> store CkSum[t-2,c,l,2]=sum_r Ck.
// This is the HBM-dominant stage (~302 MB streamed ~= 13us at 23.3 TB/s).
// ---------------------------------------------------------------------------
__global__ void ck_kernel(const float* __restrict__ fA, const float* __restrict__ fB,
                          float* __restrict__ ck01, float* __restrict__ cksum) {
    int t = blockIdx.z, c = blockIdx.y;
    int tid = threadIdx.x;
    int r = tid & 15;
    int l = blockIdx.x * 16 + (tid >> 4);
    const float* pa = fA + ((size_t)(((t * C_ + c) * L_ + l) * K_) * R_ + r) * 2;
    const float* pb = fB + ((size_t)((t * C_ + c) * K_) * L_ * R_ + (size_t)l * R_ + r) * 2;
    float accRe = 0.f, accIm = 0.f;
    for (int k = 0; k < K_; ++k) {
        float ar = pa[0], ai = pa[1];
        float br = pb[0], bi = pb[1];
        accRe += ar * br - ai * bi;
        accIm += ar * bi + ai * br;
        pa += R_ * 2;          // k-stride in factor_A
        pb += L_ * R_ * 2;     // k-stride in factor_B
    }
    if (t < 2) {
        size_t o = (size_t)((t * C_ + c) * L_ + l) * R_ * 2 + r * 2;
        ck01[o] = accRe; ck01[o + 1] = accIm;
    } else {
        // reduce over r within each 16-lane half of the wave
        for (int msk = 8; msk > 0; msk >>= 1) {
            accRe += __shfl_xor(accRe, msk, 32);
            accIm += __shfl_xor(accIm, msk, 32);
        }
        if (r == 0) {
            size_t o = (size_t)(((t - 2) * C_ + c) * L_ + l) * 2;
            cksum[o] = accRe; cksum[o + 1] = accIm;
        }
    }
}

// ---------------------------------------------------------------------------
// Kernel 3: shift[b,c,r] = sum_l cmul(mod[b,l], w_mix[l, c*R+r])
// ---------------------------------------------------------------------------
__global__ void shift_kernel(const float* __restrict__ mod, const float* __restrict__ wmix,
                             float* __restrict__ shift) {
    int r = blockIdx.x, c = blockIdx.y, b = blockIdx.z;
    int l = threadIdx.x;
    float mRe = mod[(size_t)(b * L_ + l) * 2], mIm = mod[(size_t)(b * L_ + l) * 2 + 1];
    const float* wv = wmix + (size_t)(l * (C_ * R_) + c * R_ + r) * 2;
    float wRe = wv[0], wIm = wv[1];
    float re = mRe * wRe - mIm * wIm;
    float im = mRe * wIm + mIm * wRe;
    __shared__ float sr[256], si[256];
    sr[l] = re; si[l] = im;
    __syncthreads();
    for (int st = 128; st > 0; st >>= 1) {
        if (l < st) { sr[l] += sr[l + st]; si[l] += si[l + st]; }
        __syncthreads();
    }
    if (l == 0) {
        size_t o = (size_t)((b * C_ + c) * R_ + r) * 2;
        shift[o] = sr[0]; shift[o + 1] = si[0];
    }
}

// ---------------------------------------------------------------------------
// Kernel 4: moeb[b,tt,c] = (1/(L*R)) * sum_l cmul(mod[b,l], CkSum[tt,c,l])
// (coef = z[:,2:].mean(3,4) by linearity of cmul)
// ---------------------------------------------------------------------------
__global__ void moeb_kernel(const float* __restrict__ mod, const float* __restrict__ cksum,
                            float* __restrict__ moeb) {
    int c = blockIdx.x, tt = blockIdx.y, b = blockIdx.z;
    int l = threadIdx.x;
    float mRe = mod[(size_t)(b * L_ + l) * 2], mIm = mod[(size_t)(b * L_ + l) * 2 + 1];
    const float* cs = cksum + (size_t)((tt * C_ + c) * L_ + l) * 2;
    float kRe = cs[0], kIm = cs[1];
    float re = mRe * kRe - mIm * kIm;
    float im = mRe * kIm + mIm * kRe;
    __shared__ float sr[256], si[256];
    sr[l] = re; si[l] = im;
    __syncthreads();
    for (int st = 128; st > 0; st >>= 1) {
        if (l < st) { sr[l] += sr[l + st]; si[l] += si[l + st]; }
        __syncthreads();
    }
    if (l == 0) {
        const float scale = 1.f / (float)(L_ * R_);
        size_t o = (size_t)((b * 16 + tt) * C_ + c) * 2;
        moeb[o] = sr[0] * scale; moeb[o + 1] = si[0] * scale;
    }
}

// ---------------------------------------------------------------------------
// Kernel 5: WMMA operands, pre-packed in fragment order (f16).
// interp1d is identity here (L == O0 == O1), so zz[b,c,r,o] = z[b,t,c,o,r].
//   v(dir,o,r) = cmul(space+shift, cmul(mod, Ck))
//   dir 0 (A):  siA [b,c,i, k0..31]  k<16: Re(r=k),  k>=16: Im(r=k-16)
//   dir 1 (B):  sjfr[b,c,j, k0..31]  k<16: Re(r=k),  k>=16: -Im(r=k-16)   ([Br;-Bi])
//               sjfi[b,c,j, k0..31]  k<16: Im(r=k),  k>=16:  Re(r=k-16)   ([Bi; Br])
// ---------------------------------------------------------------------------
__global__ void sisj_kernel(const float* __restrict__ mod, const float* __restrict__ ck01,
                            const float* __restrict__ shift,
                            const float* __restrict__ space_i, const float* __restrict__ space_j,
                            _Float16* __restrict__ siA, _Float16* __restrict__ sjfr,
                            _Float16* __restrict__ sjfi) {
    int r = blockIdx.x, c = blockIdx.y;
    int b = blockIdx.z >> 1, dir = blockIdx.z & 1;
    int o = threadIdx.x;
    float mRe = mod[(size_t)(b * L_ + o) * 2], mIm = mod[(size_t)(b * L_ + o) * 2 + 1];
    size_t cko = (size_t)((dir * C_ + c) * L_ + o) * R_ * 2 + r * 2;
    float kRe = ck01[cko], kIm = ck01[cko + 1];
    float zRe = mRe * kRe - mIm * kIm;
    float zIm = mRe * kIm + mIm * kRe;
    size_t sho = (size_t)((b * C_ + c) * R_ + r) * 2;
    float shRe = shift[sho], shIm = shift[sho + 1];
    const float* sp = (dir == 0 ? space_i : space_j) + (size_t)((c * R_ + r) * O0_ + o) * 2;
    float pRe = sp[0] + shRe, pIm = sp[1] + shIm;
    float vRe = pRe * zRe - pIm * zIm;
    float vIm = pRe * zIm + pIm * zRe;
    size_t rowbase = ((size_t)(b * C_ + c) * O0_ + o) * 32;   // O0_ == O1_
    if (dir == 0) {
        siA[rowbase + r]      = (_Float16)vRe;
        siA[rowbase + 16 + r] = (_Float16)vIm;
    } else {
        sjfr[rowbase + r]      = (_Float16)vRe;
        sjfr[rowbase + 16 + r] = (_Float16)(-vIm);
        sjfi[rowbase + r]      = (_Float16)vIm;
        sjfi[rowbase + 16 + r] = (_Float16)vRe;
    }
}

// ---------------------------------------------------------------------------
// Kernel 6: per (b, 16x16 tile): complex rank-16 GEMM via V_WMMA_F32_16X16X32_F16
// with real/imag packed along K=32:
//   fr = [Ar|Ai] x [Br;-Bi],  fi = [Ar|Ai] x [Bi;Br]
// Operands are pre-packed in fragment order: A = two contiguous v8h loads,
// B = one contiguous v16h load per fragment (no per-lane selects).
// Then in-register Moebius chain (8 steps) + projection + channel-sum -> out.
// One wave per tile; 8 waves/block; grid (4 j-groups, 8 i-groups, B).
// ---------------------------------------------------------------------------
__global__ void field_kernel(const _Float16* __restrict__ siA, const _Float16* __restrict__ sjfr,
                             const _Float16* __restrict__ sjfi,
                             const float* __restrict__ moeb, const float* __restrict__ inversions,
                             const float* __restrict__ proj, float* __restrict__ out) {
    int b    = blockIdx.z;
    int wid  = threadIdx.x >> 5;
    int lane = threadIdx.x & 31;
    int i0 = (blockIdx.y * 2 + (wid >> 2)) * 16;
    int j0 = (blockIdx.x * 4 + (wid & 3)) * 16;
    int m  = lane & 15;
    bool hi = lane >= 16;
    int jj = j0 + m;
    int kbase = hi ? 8 : 0;    // unused (ISA table offset 8h); A fragment uses khi
    int khi   = hi ? 16 : 0;   // B fragment: lanes 0-15 K=0..15, lanes 16-31 K=16..31

    float accOut[8];
#pragma unroll
    for (int e = 0; e < 8; ++e) accOut[e] = 0.f;

    for (int c = 0; c < C_; ++c) {
        // ---- A fragment: element e -> k = khi + e ----
        const _Float16* abase = siA + ((size_t)(b * C_ + c) * O0_ + (i0 + m)) * 32;
        v8h alo = *(const v8h*)(abase + khi);
        v8h ahi = *(const v8h*)(abase + khi + 8);
        v16h A = __builtin_shufflevector(alo, ahi, 0, 1, 2, 3, 4, 5, 6, 7,
                                                   8, 9, 10, 11, 12, 13, 14, 15);
        // ---- B fragments: element e -> k = khi + e (contiguous 32B load) ----
        const _Float16* bbase = (const _Float16*)(((size_t)(b * C_ + c) * O1_ + jj) * 32 + khi);
        v16h Bfr = *(const v16h*)(sjfr + ((size_t)(b * C_ + c) * O1_ + jj) * 32 + khi);
        v16h Bfi = *(const v16h*)(sjfi + ((size_t)(b * C_ + c) * O1_ + jj) * 32 + khi);
        (void)bbase;

        v8f fr = {}; v8f fi = {};
        fr = __builtin_amdgcn_wmma_f32_16x16x32_f16(false, A, false, Bfr, (short)0, fr, false, false);
        fi = __builtin_amdgcn_wmma_f32_16x16x32_f16(false, A, false, Bfi, (short)0, fi, false, false);

        // ---- Moebius chain on accumulators (C/D layout: e -> row, lane -> col) ----
        float invRe = inversions[c * 2], invIm = inversions[c * 2 + 1];
#pragma unroll
        for (int rt = 0; rt < RT_; ++rt) {
            const float* av = moeb + (size_t)((b * 16 + 2 * rt) * C_ + c) * 2;
            const float* dv = moeb + (size_t)((b * 16 + 2 * rt + 1) * C_ + c) * 2;
            float aRe = av[0], aIm = av[1], dRe = dv[0], dIm = dv[1];
#pragma unroll
            for (int e = 0; e < 8; ++e) {
                float fRe = fr[e], fIm = fi[e];
                float numRe = aRe * fRe - aIm * fIm + invRe;
                float numIm = aRe * fIm + aIm * fRe + invIm;
                float denRe = fRe + dRe;
                float denIm = fIm + dIm;
                float den = fmaxf(denRe * denRe + denIm * denIm, EPS_);
                fr[e] = (numRe * denRe + numIm * denIm) / den;
                fi[e] = (numIm * denRe - numRe * denIm) / den;
            }
        }
        // ---- projection, keep real part, accumulate over channels ----
#pragma unroll
        for (int e = 0; e < 8; ++e) {
            int ii = i0 + (hi ? 8 + e : e);
            const float* pv = proj + (size_t)((c * O0_ + ii) * O1_ + jj) * 2;
            accOut[e] += fr[e] * pv[0] - fi[e] * pv[1];
        }
    }
#pragma unroll
    for (int e = 0; e < 8; ++e) {
        int ii = i0 + (hi ? 8 + e : e);
        out[(size_t)(b * O0_ + ii) * O1_ + jj] = accOut[e];
    }
}

// ---------------------------------------------------------------------------
// Launcher. Workspace layout (floats then f16), ~8.9 MB total:
//   mod 4096 | Ck01 524288 | CkSum 131072 | shift 4096 | moeb 4096 f32
//   siA 2MB | sjfr 2MB | sjfi 2MB (f16, fragment-ordered)
// ---------------------------------------------------------------------------
extern "C" void kernel_launch(void* const* d_in, const int* in_sizes, int n_in,
                              void* d_out, int out_size, void* d_ws, size_t ws_size,
                              hipStream_t stream) {
    const float* x      = (const float*)d_in[0];
    const float* ctx_w  = (const float*)d_in[1];
    const float* ctx_b  = (const float*)d_in[2];
    const float* fA     = (const float*)d_in[3];
    const float* fB     = (const float*)d_in[4];
    const float* sp_i   = (const float*)d_in[5];
    const float* sp_j   = (const float*)d_in[6];
    const float* invs   = (const float*)d_in[7];
    const float* w_mix  = (const float*)d_in[8];
    const float* projs  = (const float*)d_in[9];
    float* out = (float*)d_out;

    float* ws    = (float*)d_ws;
    float* mod   = ws;                          // B*L*2          = 4096
    float* ck01  = mod + 4096;                  // 2*C*L*R*2      = 262144 (524288 reserved)
    float* cksum = ck01 + 524288;               // 16*C*L*2       = 131072
    float* shift = cksum + 131072;              // B*C*R*2        = 4096
    float* moeb  = shift + 4096;                // B*16*C*2       = 4096
    _Float16* siA  = (_Float16*)(moeb + 4096);                    // B*C*O0*32 halves
    _Float16* sjfr = siA  + (size_t)B_ * C_ * O0_ * 32;           // B*C*O1*32 halves
    _Float16* sjfi = sjfr + (size_t)B_ * C_ * O1_ * 32;           // B*C*O1*32 halves

    mod_kernel  <<<dim3(2 * L_, B_),      256, 0, stream>>>(x, ctx_w, ctx_b, mod);
    ck_kernel   <<<dim3(L_ / 16, C_, T_), 256, 0, stream>>>(fA, fB, ck01, cksum);
    shift_kernel<<<dim3(R_, C_, B_),      256, 0, stream>>>(mod, w_mix, shift);
    moeb_kernel <<<dim3(C_, 16, B_),      256, 0, stream>>>(mod, cksum, moeb);
    sisj_kernel <<<dim3(R_, C_, 2 * B_),  256, 0, stream>>>(mod, ck01, shift, sp_i, sp_j,
                                                            siA, sjfr, sjfi);
    field_kernel<<<dim3(4, 8, B_),        256, 0, stream>>>(siA, sjfr, sjfi, moeb, invs, projs, out);
}
